// CrystalGraphConvNet_66073776881701
// MI455X (gfx1250) — compile-verified, matrix operand order checked by reading, not run on hardware
//
#include <hip/hip_runtime.h>
#include <hip/hip_bf16.h>

typedef __attribute__((ext_vector_type(16))) __bf16 v16bf;
typedef __attribute__((ext_vector_type(8)))  __bf16 v8bf;
typedef __attribute__((ext_vector_type(8)))  float  v8f;

#define N_ATOMS 100000
#define M_NBR   12
#define NFE     41
#define AF      64
#define NROWS   (N_ATOMS * M_NBR)   // 1,200,000
#define KIN     (2*AF + NFE)        // 169
#define KPAD    192                 // 6 chunks of 32
#define COLS    128                 // 2*AF
#define NCONV   3
#define NCRYS   100
#define APC     1000
#define HF      128
#define HID     64
#define BN_EPS  1e-5f

#define ROWS_PER_BLOCK 128          // 8 waves x 16 rows
#define SA_STRIDE 200               // 192 + pad (400B row: conflict-free b128)
#define SB_STRIDE 200               // transposed W tile, same stride
#define SO_STRIDE 136               // epilogue staging stride (bf16)

__device__ __forceinline__ float softplusf(float x) {
    // log1p(exp(x)) stabilized: max(x,0) + log1p(exp(-|x|))
    return fmaxf(x, 0.f) + log1pf(expf(-fabsf(x)));
}
__device__ __forceinline__ float sigmoidf(float x) {
    return 1.f / (1.f + expf(-x));
}

// ---------------------------------------------------------------- embedding
__global__ __launch_bounds__(256) void cgcnn_embed(
    const int* __restrict__ atomNum, const float* __restrict__ emb,
    float* __restrict__ fea)
{
    int gid = blockIdx.x * 256 + threadIdx.x;   // over N_ATOMS*AF
    int n = gid >> 6, f = gid & 63;
    fea[gid] = emb[atomNum[n] * AF + f];
}

// -------------------------------------------------- conv GEMM (bf16 WMMA)
// rows = (atom, nbr) pairs; row features = [self(64) | nbr(64) | edge(41) | pad]
// out: gated[row][0..127] (bf16) + per-channel sum/sumsq for BN1
__global__ __launch_bounds__(256) void cgcnn_gemm_wmma(
    const float* __restrict__ feaIn,   // (N, 64)
    const float* __restrict__ nbrFea,  // (N, 12, 41)
    const int*   __restrict__ nbrIdx,  // (N, 12)
    const float* __restrict__ W,       // (169, 128) this layer
    const float* __restrict__ bias,    // (128,)
    __bf16*      __restrict__ gated,   // (N*12, 128)
    float* __restrict__ gSum, float* __restrict__ gSumSq)
{
    __shared__ __bf16 sA[ROWS_PER_BLOCK * SA_STRIDE];   // 50 KB (reused for epilogue)
    __shared__ __bf16 sBt[COLS * SB_STRIDE];            // 50 KB, transposed: [n][k]
    __shared__ float  sSum[COLS];
    __shared__ float  sSumSq[COLS];

    const int tid = threadIdx.x;
    const int rowBase = blockIdx.x * ROWS_PER_BLOCK;

    if (tid < COLS) { sSum[tid] = 0.f; sSumSq[tid] = 0.f; }

    // stage weights -> LDS bf16, TRANSPOSED (fragment loads become b128),
    // zero-pad K = 169..191. Global reads stay coalesced over n.
    for (int idx = tid; idx < KPAD * COLS; idx += 256) {
        int k = idx >> 7;
        int nn = idx & 127;
        sBt[nn * SB_STRIDE + k] = (k < KIN) ? (__bf16)W[k * COLS + nn] : (__bf16)0.f;
    }
    // stage 128 concat rows -> LDS bf16
    for (int idx = tid; idx < ROWS_PER_BLOCK * KPAD; idx += 256) {
        int r = idx / KPAD;
        int f = idx - r * KPAD;
        int row = rowBase + r;
        int n = row / M_NBR;
        float v;
        if (f < AF)            v = feaIn[n * AF + f];
        else if (f < 2 * AF)   v = feaIn[nbrIdx[row] * AF + (f - AF)];
        else if (f < KIN)      v = nbrFea[(size_t)row * NFE + (f - 2 * AF)];
        else                   v = 0.f;
        sA[r * SA_STRIDE + f] = (__bf16)v;
    }
    __syncthreads();

    const int wave = tid >> 5;
    const int lane = tid & 31;
    const int l16  = lane & 15;
    const int hi   = lane >> 4;

    v8f acc[8] = {};
    const int arow = wave * 16 + l16;

    #pragma unroll
    for (int kc = 0; kc < KPAD / 32; ++kc) {
        const int kb = kc * 32 + hi * 8;        // 16B-aligned fragment base
        // A fragment (ISA 7.12.2 16-bit A layout): two contiguous 8xbf16 runs
        v8bf alo = *(const v8bf*)(sA + arow * SA_STRIDE + kb);
        v8bf ahi = *(const v8bf*)(sA + arow * SA_STRIDE + kb + 16);
        v16bf a = __builtin_shufflevector(alo, ahi,
                    0,1,2,3,4,5,6,7,8,9,10,11,12,13,14,15);
        #pragma unroll
        for (int nt = 0; nt < 8; ++nt) {
            const int col = nt * 16 + l16;
            // B fragment: same shape thanks to transposed sBt
            v8bf blo = *(const v8bf*)(sBt + col * SB_STRIDE + kb);
            v8bf bhi = *(const v8bf*)(sBt + col * SB_STRIDE + kb + 16);
            v16bf b = __builtin_shufflevector(blo, bhi,
                        0,1,2,3,4,5,6,7,8,9,10,11,12,13,14,15);
            acc[nt] = __builtin_amdgcn_wmma_f32_16x16x32_bf16(
                false, a, false, b, (short)0, acc[nt], false, false);
        }
    }

    __syncthreads();   // everyone done reading sA/sBt; reuse sA as out-staging

    // epilogue: +bias, BN1 stats, stage bf16 tile in LDS (stride SO_STRIDE)
    #pragma unroll
    for (int nt = 0; nt < 8; ++nt) {
        const int col = nt * 16 + l16;
        const float bv = bias[col];
        float s = 0.f, sq = 0.f;
        #pragma unroll
        for (int j = 0; j < 8; ++j) {
            int rl = j + hi * 8;                 // C/D layout: VGPR j, half -> row
            float v = acc[nt][j] + bv;
            sA[(wave * 16 + rl) * SO_STRIDE + col] = (__bf16)v;
            s  += v;
            sq += v * v;
        }
        atomicAdd(&sSum[col], s);
        atomicAdd(&sSumSq[col], sq);
    }
    __syncthreads();

    // cooperative, fully-coalesced bf16 tile store (16B per thread per iter)
    for (int idx = tid; idx < ROWS_PER_BLOCK * (COLS / 8); idx += 256) {
        int r  = idx >> 4;                       // 16 chunks of 8 bf16 per row
        int ch = idx & 15;
        v8bf vv = *(const v8bf*)(sA + r * SO_STRIDE + ch * 8);
        *(v8bf*)(gated + (size_t)(rowBase + r) * COLS + ch * 8) = vv;
    }
    if (tid < COLS) {
        atomicAdd(&gSum[tid],   sSum[tid]);
        atomicAdd(&gSumSq[tid], sSumSq[tid]);
    }
}

// ----------------------------------------------- BN stats -> scale/shift
__global__ void bn_finalize(const float* __restrict__ sum,
                            const float* __restrict__ sumsq,
                            const float* __restrict__ g,
                            const float* __restrict__ b,
                            float* __restrict__ scale,
                            float* __restrict__ shift,
                            int nch, float invCnt)
{
    int t = threadIdx.x;
    if (t < nch) {
        float mean = sum[t] * invCnt;
        float var  = fmaxf(sumsq[t] * invCnt - mean * mean, 0.f);
        float sc   = g[t] * rsqrtf(var + BN_EPS);
        scale[t] = sc;
        shift[t] = b[t] - mean * sc;
    }
}

// ----------------------- BN1 affine + sigmoid*softplus + sum over M + BN2 stats
__global__ __launch_bounds__(256) void cgcnn_gate_reduce(
    const __bf16* __restrict__ gated,
    const float* __restrict__ scale1, const float* __restrict__ shift1,
    float* __restrict__ summed,
    float* __restrict__ gSum2, float* __restrict__ gSumSq2)
{
    __shared__ float ls[AF], lq[AF];
    const int tid = threadIdx.x;
    if (tid < AF) { ls[tid] = 0.f; lq[tid] = 0.f; }
    __syncthreads();

    const int gid = blockIdx.x * 256 + tid;     // over N_ATOMS*AF
    const int n = gid >> 6;
    const int f = gid & 63;
    const float scF = scale1[f],      shF = shift1[f];
    const float scC = scale1[AF + f], shC = shift1[AF + f];

    float acc = 0.f;
    #pragma unroll
    for (int m = 0; m < M_NBR; ++m) {
        size_t base = ((size_t)n * M_NBR + m) * COLS;
        float gf = (float)gated[base + f];
        float gc = (float)gated[base + AF + f];
        acc += sigmoidf(gf * scF + shF) * softplusf(gc * scC + shC);
    }
    summed[gid] = acc;
    atomicAdd(&ls[f], acc);
    atomicAdd(&lq[f], acc * acc);
    __syncthreads();
    if (tid < AF) {
        atomicAdd(&gSum2[tid],   ls[tid]);
        atomicAdd(&gSumSq2[tid], lq[tid]);
    }
}

// ------------------------------------ BN2 affine + residual softplus
__global__ __launch_bounds__(256) void cgcnn_update(
    const float* __restrict__ feaIn, const float* __restrict__ summed,
    const float* __restrict__ scale2, const float* __restrict__ shift2,
    float* __restrict__ feaOut)
{
    int gid = blockIdx.x * 256 + threadIdx.x;
    int f = gid & 63;
    feaOut[gid] = softplusf(feaIn[gid] + summed[gid] * scale2[f] + shift2[f]);
}

// ------------------------------------------------------- crystal pooling
__global__ __launch_bounds__(256) void cgcnn_pool(
    const float* __restrict__ fea, const int* __restrict__ cidx,
    float* __restrict__ crys)
{
    __shared__ float red[256];
    int c = blockIdx.x, tid = threadIdx.x;
    int f = tid & 63, part = tid >> 6;
    float s = 0.f;
    for (int a = part; a < APC; a += 4)
        s += fea[(size_t)cidx[c * APC + a] * AF + f];
    red[tid] = s;
    __syncthreads();
    if (part == 0)
        crys[c * AF + f] =
            (red[f] + red[64 + f] + red[128 + f] + red[192 + f]) * (1.f / APC);
}

// ------------------------------------------------------------ output MLP
__global__ __launch_bounds__(128) void cgcnn_mlp(
    const float* __restrict__ crys,
    const float* __restrict__ Wfc, const float* __restrict__ bfc,
    const float* __restrict__ Wout, const float* __restrict__ bout,
    float* __restrict__ out)
{
    __shared__ float sIn[AF], sHid[HF];
    int c = blockIdx.x, t = threadIdx.x;
    if (t < AF) sIn[t] = softplusf(crys[c * AF + t]);
    __syncthreads();
    float h = bfc[t];
    #pragma unroll 8
    for (int k = 0; k < AF; ++k) h += sIn[k] * Wfc[k * HF + t];
    sHid[t] = softplusf(h);
    __syncthreads();
    if (t < HID) {
        float o = bout[t];
        #pragma unroll 8
        for (int k = 0; k < HF; ++k) o += sHid[k] * Wout[k * HID + t];
        out[c * HID + t] = o;
    }
}

// ---------------------------------------------------------------- launch
extern "C" void kernel_launch(void* const* d_in, const int* in_sizes, int n_in,
                              void* d_out, int out_size, void* d_ws, size_t ws_size,
                              hipStream_t stream)
{
    (void)in_sizes; (void)n_in; (void)out_size; (void)ws_size;
    const int*   atomNum = (const int*)  d_in[0];
    const float* nbrFea  = (const float*)d_in[1];
    const int*   nbrIdx  = (const int*)  d_in[2];
    const int*   cidx    = (const int*)  d_in[3];
    const float* emb     = (const float*)d_in[4];
    const float* Wfull   = (const float*)d_in[5];
    const float* bfull   = (const float*)d_in[6];
    const float* bn1g    = (const float*)d_in[7];
    const float* bn1b    = (const float*)d_in[8];
    const float* bn2g    = (const float*)d_in[9];
    const float* bn2b    = (const float*)d_in[10];
    const float* Wfc     = (const float*)d_in[11];
    const float* bfc     = (const float*)d_in[12];
    const float* Wout    = (const float*)d_in[13];
    const float* bout    = (const float*)d_in[14];
    float* out = (float*)d_out;

    char* ws = (char*)d_ws;
    size_t off = 0;
    auto take = [&](size_t bytes) -> char* {
        char* p = ws + off;
        off = (off + bytes + 255) & ~(size_t)255;
        return p;
    };
    float*  fea0   = (float*) take((size_t)N_ATOMS * AF * 4);
    float*  fea1   = (float*) take((size_t)N_ATOMS * AF * 4);
    float*  summed = (float*) take((size_t)N_ATOMS * AF * 4);
    float*  crys   = (float*) take((size_t)NCRYS * AF * 4);
    float*  sum1   = (float*) take(2 * COLS * 4);   float* sumsq1 = sum1 + COLS;
    float*  scale1 = (float*) take(2 * COLS * 4);   float* shift1 = scale1 + COLS;
    float*  sum2   = (float*) take(2 * AF * 4);     float* sumsq2 = sum2 + AF;
    float*  scale2 = (float*) take(2 * AF * 4);     float* shift2 = scale2 + AF;
    __bf16* gated  = (__bf16*)take((size_t)NROWS * COLS * 2);

    cgcnn_embed<<<(N_ATOMS * AF) / 256, 256, 0, stream>>>(atomNum, emb, fea0);

    float* fin = fea0; float* fout = fea1;
    for (int i = 0; i < NCONV; ++i) {
        hipMemsetAsync(sum1, 0, 2 * COLS * 4, stream);
        hipMemsetAsync(sum2, 0, 2 * AF * 4, stream);
        cgcnn_gemm_wmma<<<NROWS / ROWS_PER_BLOCK, 256, 0, stream>>>(
            fin, nbrFea, nbrIdx,
            Wfull + (size_t)i * KIN * COLS, bfull + i * COLS,
            gated, sum1, sumsq1);
        bn_finalize<<<1, 128, 0, stream>>>(sum1, sumsq1, bn1g + i * COLS,
                                           bn1b + i * COLS, scale1, shift1,
                                           COLS, 1.f / (float)NROWS);
        cgcnn_gate_reduce<<<(N_ATOMS * AF) / 256, 256, 0, stream>>>(
            gated, scale1, shift1, summed, sum2, sumsq2);
        bn_finalize<<<1, 128, 0, stream>>>(sum2, sumsq2, bn2g + i * AF,
                                           bn2b + i * AF, scale2, shift2,
                                           AF, 1.f / (float)N_ATOMS);
        cgcnn_update<<<(N_ATOMS * AF) / 256, 256, 0, stream>>>(
            fin, summed, scale2, shift2, fout);
        float* t = fin; fin = fout; fout = t;
    }

    cgcnn_pool<<<NCRYS, 256, 0, stream>>>(fin, cidx, crys);
    cgcnn_mlp<<<NCRYS, HF, 0, stream>>>(crys, Wfc, bfc, Wout, bout, out);
}
